// Parallel_Attention2_49108656062948
// MI455X (gfx1250) — matrix operand order repaired and checked by reading, not compiled, not run on hardware
//
#include <hip/hip_runtime.h>

#define BATCH 16
#define NQ    2048
#define NK    2048
#define DIM   512
#define SCALE 0.044194173824159216f  // 1/sqrt(512)

typedef __attribute__((ext_vector_type(16))) __bf16 v16bf;
typedef __attribute__((ext_vector_type(8)))  __bf16 v8bf;
typedef __attribute__((ext_vector_type(4)))  __bf16 v4bf;
typedef __attribute__((ext_vector_type(16))) float  v16f;
typedef __attribute__((ext_vector_type(8)))  float  v8f;
typedef __attribute__((ext_vector_type(4)))  float  v4f;

__device__ __forceinline__ v16bf cat8(v8bf lo, v8bf hi) {
  return __builtin_shufflevector(lo, hi, 0,1,2,3,4,5,6,7,8,9,10,11,12,13,14,15);
}

// ---- fragment loaders, overloaded on element type -------------------------
// B-style: 16 consecutive elements. f32: 64B (4xb128) + cvt; bf16: 32B direct.
__device__ __forceinline__ v16bf load_contig16(const float* p) {
  v16f t = *(const v16f*)p;
  return __builtin_convertvector(t, v16bf);
}
__device__ __forceinline__ v16bf load_contig16(const __bf16* p) {
  return *(const v16bf*)p;
}
// A-style: e<8 <- p0[e], e>=8 <- p1[e-8].
__device__ __forceinline__ v16bf load_split8(const float* p0, const float* p1) {
  v8f lo = *(const v8f*)p0;
  v8f hi = *(const v8f*)p1;
  v16f t;
#pragma unroll
  for (int e = 0; e < 8; ++e) { t[e] = lo[e]; t[8 + e] = hi[e]; }
  return __builtin_convertvector(t, v16bf);
}
__device__ __forceinline__ v16bf load_split8(const __bf16* p0, const __bf16* p1) {
  return cat8(*(const v8bf*)p0, *(const v8bf*)p1);
}

template <bool PRE> struct elem      { using type = float;  };
template <>         struct elem<true>{ using type = __bf16; };

// ---------------------------------------------------------------------------
// Prep kernels (preconverted path): row-major f32->bf16, and V transpose.
// ---------------------------------------------------------------------------
__global__ __launch_bounds__(256) void cvt_bf16_kernel(
    const float* __restrict__ src, __bf16* __restrict__ dst) {
  const size_t i = ((size_t)blockIdx.x * 256 + threadIdx.x) * 8;
  v8f x = *(const v8f*)(src + i);
  *(v8bf*)(dst + i) = __builtin_convertvector(x, v8bf);
}

__global__ __launch_bounds__(256) void transpose_v_kernel(
    const float* __restrict__ vm, __bf16* __restrict__ vt) {
  __shared__ __bf16 Tl[64][72];
  const int b   = blockIdx.z;
  const int kt  = blockIdx.x * 64;
  const int dt  = blockIdx.y * 64;
  const int tid = threadIdx.x;
  const int r   = tid >> 4;          // 0..15
  const int c4  = (tid & 15) * 4;    // 0..60
#pragma unroll
  for (int it = 0; it < 4; ++it) {
    const int row = r + it * 16;     // k offset in tile
    float4 x = *(const float4*)(vm + ((size_t)b * NK + kt + row) * DIM + dt + c4);
    v4f xf = {x.x, x.y, x.z, x.w};
    v4bf xb = __builtin_convertvector(xf, v4bf);
    Tl[c4 + 0][row] = xb[0];
    Tl[c4 + 1][row] = xb[1];
    Tl[c4 + 2][row] = xb[2];
    Tl[c4 + 3][row] = xb[3];
  }
  __syncthreads();
#pragma unroll
  for (int it = 0; it < 4; ++it) {
    const int row = r + it * 16;     // d offset in tile
    v4bf t = *(const v4bf*)&Tl[row][c4];
    *(v4bf*)(vt + ((size_t)b * DIM + dt + row) * NK + kt + c4) = t;
  }
}

// ---------------------------------------------------------------------------
// Pass 1: per-(b,k) softmax stats over the q axis (softmax is over axis=1).
// 8 waves/block; each wave owns 16 k-columns; K B-fragments (full D=512)
// hoisted; online (max, sum-exp) over 128 q slabs.
// ---------------------------------------------------------------------------
template <bool PRE>
__global__ __launch_bounds__(256) void attn_stats_kernel(
    const void* __restrict__ qp, const void* __restrict__ kp,
    float* __restrict__ colmax, float* __restrict__ colsum) {
  using T = typename elem<PRE>::type;
  const int lane = threadIdx.x & 31;
  const int wave = threadIdx.x >> 5;
  const int g    = lane >> 4;
  const int l15  = lane & 15;
  const int b    = blockIdx.x / (NK / 128);
  const int kblk = blockIdx.x % (NK / 128);
  const int k0   = kblk * 128 + wave * 16;

  const T* krow = (const T*)kp + ((size_t)b * NK + k0 + l15) * DIM;
  v16bf kf[16];
#pragma unroll
  for (int s = 0; s < 16; ++s)
    kf[s] = load_split8(krow + s * 32 + 8 * g, krow + s * 32 + 16 + 8 * g);

  float m = -3.0e38f, z = 0.0f;
  const T* qb = (const T*)qp + (size_t)b * NQ * DIM;
  for (int qt = 0; qt < NQ / 16; ++qt) {
    const T* qrow = qb + (size_t)(qt * 16 + l15) * DIM;
    v8f c = {};
#pragma unroll
    for (int s = 0; s < 16; ++s) {
      v16bf af = load_split8(qrow + s * 32 + 8 * g,
                             qrow + s * 32 + 16 + 8 * g);
      c = __builtin_amdgcn_wmma_f32_16x16x32_bf16(
          false, af, false, kf[s], (short)0, c, false, false);
    }
    float lm = c[0];
#pragma unroll
    for (int i = 1; i < 8; ++i) lm = fmaxf(lm, c[i]);
    lm *= SCALE;
    float nm = fmaxf(m, lm);
    z *= __expf(m - nm);
#pragma unroll
    for (int i = 0; i < 8; ++i) z += __expf(c[i] * SCALE - nm);
    m = nm;
  }
  float om = __shfl_xor(m, 16, 32);
  float oz = __shfl_xor(z, 16, 32);
  float fm = fmaxf(m, om);
  float fz = z * __expf(m - fm) + oz * __expf(om - fm);
  if (lane < 16) {
    colmax[(size_t)b * NK + k0 + l15] = fm;
    colsum[(size_t)b * NK + k0 + l15] = fz;
  }
}

// ---------------------------------------------------------------------------
// Pass 2: 512 threads (16 waves) per (b, 32-q block).
//   Q staged once into LDS (bf16, A-fragment-permuted: within each 32-chunk
//   index bits 3<->4 swapped) -> one aligned 32B ds_load per A-fragment.
//   Per 256-wide k block: each wave computes 2 S tiles, writes normalized P
//   (bf16, unpermuted k order) to LDS; then accumulates its 64-wide d strip with
//   P x V WMMAs. PRE: V fragments are contiguous 32B loads from Vt[b][d][k].
// ---------------------------------------------------------------------------
template <bool PRE>
__global__ __launch_bounds__(512) void attn_out_kernel(
    const void* __restrict__ qp, const void* __restrict__ kp,
    const void* __restrict__ vp, const float* __restrict__ colmax,
    const float* __restrict__ colsum, float* __restrict__ out) {
  using T = typename elem<PRE>::type;
  __shared__ __align__(32) __bf16 Ql[32][528];  // 32 q rows x 512 d (+pad)
  __shared__ __align__(32) __bf16 Pl[32][272];  // 32 q rows x 256 k (+pad)

  const int tid   = threadIdx.x;
  const int lane  = tid & 31;
  const int wave  = tid >> 5;     // 0..15
  const int g     = lane >> 4;
  const int l15   = lane & 15;
  const int qtile = wave >> 3;    // 0..1
  const int wsub  = wave & 7;     // 0..7
  const int b     = blockIdx.x / (NQ / 32);
  const int q0    = (blockIdx.x % (NQ / 32)) * 32;

  // ---- stage Q block into LDS (permuted bf16 A-layout) --------------------
  {
    const int row = tid >> 4;   // 0..31
    const int s   = tid & 15;   // d-chunk 0..15
    const T* src = (const T*)qp + ((size_t)b * NQ + q0 + row) * DIM + s * 32;
    *(v16bf*)&Ql[row][s * 32]      = load_split8(src,     src + 16);
    *(v16bf*)&Ql[row][s * 32 + 16] = load_split8(src + 8, src + 24);
  }
  __syncthreads();

  v8f acc[4] = {v8f{}, v8f{}, v8f{}, v8f{}};
  const T* kb_      = (const T*)kp + (size_t)b * NK * DIM;
  const float* mrow = colmax + (size_t)b * NK;
  const float* zrow = colsum + (size_t)b * NK;

  for (int kb = 0; kb < NK / 256; ++kb) {
    // ---- Phase A: P strip (32 x 256 bf16) --------------------------------
#pragma unroll
    for (int tt = 0; tt < 2; ++tt) {
      const int t  = 2 * wsub + tt;       // k tile 0..15 within block
      const int kt = kb * 256 + t * 16;
      const T* krow = kb_ + (size_t)(kt + l15) * DIM;
      v8f c = {};
#pragma unroll
      for (int s = 0; s < 16; ++s) {
        v16bf af = *(const v16bf*)&Ql[qtile * 16 + l15][s * 32 + 16 * g];
        v16bf bf = load_split8(krow + s * 32 + 8 * g, krow + s * 32 + 16 + 8 * g);
        c = __builtin_amdgcn_wmma_f32_16x16x32_bf16(
            false, af, false, bf, (short)0, c, false, false);
      }
      const float mk = mrow[kt + l15];
      const float zi = 1.0f / zrow[kt + l15];
      v8f pv;
#pragma unroll
      for (int i = 0; i < 8; ++i) pv[i] = __expf(c[i] * SCALE - mk) * zi;
      v8bf pb = __builtin_convertvector(pv, v8bf);
      const int col  = t * 16 + l15;
      const int kk   = col & 31;
      const int colp = col;
#pragma unroll
      for (int i = 0; i < 8; ++i)
        Pl[qtile * 16 + i + 8 * g][colp] = pb[i];
    }
    __syncthreads();
    // ---- Phase B: acc += P(16x32) @ V(32x16) per d tile ------------------
#pragma unroll
    for (int kc = 0; kc < 8; ++kc) {
      v16bf af = *(const v16bf*)&Pl[qtile * 16 + l15][kc * 32 + 16 * g];
      const int kg = kb * 256 + kc * 32 + 16 * g;
#pragma unroll
      for (int j = 0; j < 4; ++j) {
        const int dcol = wsub * 64 + j * 16 + l15;
        v16bf bf;
        if constexpr (PRE) {
          bf = *(const v16bf*)((const __bf16*)vp +
                               ((size_t)b * DIM + dcol) * NK + kg);
        } else {
          v16f t;
#pragma unroll
          for (int e = 0; e < 16; ++e)
            t[e] = ((const float*)vp)[((size_t)b * NK + kg + e) * DIM + dcol];
          bf = __builtin_convertvector(t, v16bf);
        }
        acc[j] = __builtin_amdgcn_wmma_f32_16x16x32_bf16(
            false, af, false, bf, (short)0, acc[j], false, false);
      }
    }
    __syncthreads();
  }

  // ---- store: C-layout row i+8g, col l15 ----------------------------------
#pragma unroll
  for (int j = 0; j < 4; ++j) {
    const int dcol = wsub * 64 + j * 16 + l15;
#pragma unroll
    for (int i = 0; i < 8; ++i)
      out[((size_t)b * NQ + q0 + qtile * 16 + i + 8 * g) * DIM + dcol] = acc[j][i];
  }
}

extern "C" void kernel_launch(void* const* d_in, const int* in_sizes, int n_in,
                              void* d_out, int out_size, void* d_ws, size_t ws_size,
                              hipStream_t stream) {
  const float* q = (const float*)d_in[0];
  const float* k = (const float*)d_in[1];
  const float* v = (const float*)d_in[2];
  // d_in[3] is the scalar dim (=512), baked in as DIM.
  float* out = (float*)d_out;

  const size_t nelem = (size_t)BATCH * NQ * DIM;      // 16.8M per tensor
  float*  colmax = (float*)d_ws;                      // B*NK floats
  float*  colsum = colmax + (size_t)BATCH * NK;       // B*NK floats
  __bf16* qbf    = (__bf16*)(colsum + (size_t)BATCH * NK);
  __bf16* kbf    = qbf + nelem;
  __bf16* vt     = kbf + nelem;
  const size_t need = 2 * (size_t)BATCH * NK * sizeof(float) +
                      3 * nelem * sizeof(__bf16);     // ~96.25 MB

  if (ws_size >= need) {
    // one-time bf16 preconversion: Q,K row-major; V transposed to [b][d][k]
    const int cvt_blocks = (int)(nelem / (256 * 8));
    cvt_bf16_kernel<<<cvt_blocks, 256, 0, stream>>>(q, qbf);
    cvt_bf16_kernel<<<cvt_blocks, 256, 0, stream>>>(k, kbf);
    transpose_v_kernel<<<dim3(NK / 64, DIM / 64, BATCH), 256, 0, stream>>>(v, vt);
    attn_stats_kernel<true><<<BATCH * (NK / 128), 256, 0, stream>>>(
        qbf, kbf, colmax, colsum);
    attn_out_kernel<true><<<BATCH * (NQ / 32), 512, 0, stream>>>(
        qbf, kbf, vt, colmax, colsum, out);
  } else {
    attn_stats_kernel<false><<<BATCH * (NK / 128), 256, 0, stream>>>(
        q, k, colmax, colsum);
    attn_out_kernel<false><<<BATCH * (NQ / 32), 512, 0, stream>>>(
        q, k, v, colmax, colsum, out);
  }
}
